// MultiheadAttention_38122129719535
// MI455X (gfx1250) — matrix operand order repaired and checked by reading, not compiled, not run on hardware
//
#include <hip/hip_runtime.h>
#include <hip/hip_bf16.h>

#define NB 4
#define NS 2048
#define ND 128
#define NH 8
#define NN (NH * ND) /* 1024 */

#define AS1 __attribute__((address_space(1)))
#define AS3 __attribute__((address_space(3)))

#define HAS_ASYNC_LDS (__has_builtin(__builtin_amdgcn_global_load_async_to_lds_b128) && \
                       __has_builtin(__builtin_amdgcn_s_wait_asynccnt))

typedef _Float16 half16_t __attribute__((ext_vector_type(16)));
typedef _Float16 half8_t  __attribute__((ext_vector_type(8)));
typedef float    float8_t __attribute__((ext_vector_type(8)));
typedef int      int4v    __attribute__((ext_vector_type(4)));

static __device__ inline float8_t wmma_f16(half16_t a, half16_t b, float8_t c) {
  // D(16x16,f32) = A(16x32,f16) * B(32x16,f16) + C
  return __builtin_amdgcn_wmma_f32_16x16x32_f16(
      /*neg_a=*/false, a, /*neg_b=*/false, b,
      /*c_mod=*/(short)0, c, /*reuse_a=*/false, /*reuse_b=*/false);
}

// 16-bit A-matrix 16x32 layout (contiguous, matches B): lane half hf (0:lanes0-15, 1:lanes16-31),
// VGPR pair v holds K = {base, base+1} with base = 16*hf + 2*v:
static __device__ inline int a_pair_base(int v, int hf) {
  return hf * 16 + 2 * v;
}

static __device__ inline half16_t cat8(half8_t lo, half8_t hi) {
  half16_t r;
#pragma unroll
  for (int i = 0; i < 8; ++i) { r[i] = lo[i]; r[i + 8] = hi[i]; }
  return r;
}

// B-matrix 32x16 f16 fragment when the 16 needed K-values are contiguous for
// this lane: b[e] = p[e]  (two 128-bit loads)
static __device__ inline half16_t load_b_contig(const _Float16* p) {
  const half8_t* v = (const half8_t*)p;
  return cat8(v[0], v[1]);
}

// 16-byte global -> LDS copy: async (no VGPR round trip, ASYNCcnt-tracked)
// when the toolchain exposes the gfx1250 builtin, else reg staging.
// Builtin signature (from hipcc diagnostic): param0 = int4 AS1* (non-const).
static __device__ inline void cp16_g2l(const _Float16* g, _Float16* l) {
#if HAS_ASYNC_LDS
  __builtin_amdgcn_global_load_async_to_lds_b128(
      (AS1 int4v*)(unsigned long long)g,
      (AS3 int4v*)(unsigned int)(unsigned long long)l, /*offset=*/0, /*cpol=*/0);
#else
  *(half8_t*)l = *(const half8_t*)g;
#endif
}

// ---------------------------------------------------------------------------
// Projection: out = X[M=NB*NS, 128] @ W[128, 1024] + bias, emitted as f16.
// transposed==0 : out[b][h][s][d]   (Q, K)
// transposed==1 : out[b][h][d][s]   (V — attention stages V^T contiguously)
// One wave per 16x16 tile, 4x wmma_f32_16x16x32_f16 over K=128.
// ---------------------------------------------------------------------------
__global__ void __launch_bounds__(128) proj_kernel(
    const float* __restrict__ X, const float* __restrict__ W,
    const float* __restrict__ bias, _Float16* __restrict__ out, int transposed) {
  const int lane = threadIdx.x & 31;
  const int wid  = threadIdx.x >> 5;
  const int id   = blockIdx.x * 4 + wid;
  const int MT   = (NB * NS) / 16; // 512 row tiles
  const int mt = id % MT;
  const int nt = id / MT;
  const int m0 = mt * 16, n0 = nt * 16;
  const int hf = lane >> 4, l16 = lane & 15;

  const float* arow = X + (size_t)(m0 + l16) * ND;

  float8_t acc = {};
#pragma unroll
  for (int c = 0; c < 4; ++c) {
    half16_t a;
#pragma unroll
    for (int v = 0; v < 8; ++v) {
      int k = c * 32 + a_pair_base(v, hf);
      a[2 * v]     = (_Float16)arow[k];
      a[2 * v + 1] = (_Float16)arow[k + 1];
    }
    half16_t bm;
#pragma unroll
    for (int e = 0; e < 16; ++e)
      bm[e] = (_Float16)W[(size_t)(c * 32 + hf * 16 + e) * NN + n0 + l16];
    acc = wmma_f16(a, bm, acc);
  }

  const float bv = bias[n0 + l16];
  const int h = n0 / ND;
  const int d = (n0 % ND) + l16;
#pragma unroll
  for (int r = 0; r < 8; ++r) {
    int m = m0 + r + hf * 8; // global row in [0, NB*NS)
    int bb = m / NS, s = m % NS;
    _Float16 val = (_Float16)(acc[r] + bv);
    if (!transposed)
      out[(((size_t)bb * NH + h) * NS + s) * ND + d] = val;
    else
      out[(((size_t)bb * NH + h) * ND + d) * NS + s] = val;
  }
}

// ---------------------------------------------------------------------------
// Flash-style attention with |score| softmax; double-buffered async-staged
// K/V tiles in LDS shared by 8 waves (128 q rows per block of one (b,h)).
// Q,K : f16 [B,H,S,128] row-major;  Vt : f16 [B,H,128,S].
// ---------------------------------------------------------------------------
#define KROWPAD 8 /* halves; row stride 272B -> lane rows hit distinct banks */
#define VROWPAD 8 /* halves; row stride 80B */

__global__ void __launch_bounds__(256) attn_kernel(
    const _Float16* __restrict__ Qf, const _Float16* __restrict__ Kf,
    const _Float16* __restrict__ Vt, float* __restrict__ out) {
  __shared__ _Float16 Ks[2][32][ND + KROWPAD]; // K tile, row-major [kv][d]
  __shared__ _Float16 Vs[2][ND][32 + VROWPAD]; // V tile, [d][kv] (from Vt)
  __shared__ _Float16 Ps[8][16 * 32];          // per-wave P staging (C->A)

  const int tid  = threadIdx.x;
  const int lane = tid & 31;
  const int wid  = tid >> 5;
  const int qblk = blockIdx.x & (NS / 128 - 1); // 16 q blocks per head
  const int bh   = blockIdx.x >> 4;             // [0, NB*NH)
  const int h = bh % NH, b = bh / NH;
  const int q0 = qblk * 128 + wid * 16;
  const int hf = lane >> 4, l16 = lane & 15;

  const _Float16* Qb = Qf + ((size_t)(b * NH + h) * NS) * ND;
  const _Float16* Kb = Kf + ((size_t)(b * NH + h) * NS) * ND;
  const _Float16* Vb = Vt + ((size_t)(b * NH + h) * ND) * NS;
  _Float16* pl = &Ps[wid][0];

  // staging coordinates (each thread moves 2x16B of K and 2x16B of V = 4 ops)
  const int kr = tid >> 3, kc = (tid & 7) * 16; // K: 32 rows x 128 halves
  const int vr = tid >> 1, vc = (tid & 1) * 16; // V: 128 rows x 32 halves

  auto stage = [&](int buf, int kv0) {
    const _Float16* gk = Kb + (size_t)(kv0 + kr) * ND + kc;
    cp16_g2l(gk,     &Ks[buf][kr][kc]);
    cp16_g2l(gk + 8, &Ks[buf][kr][kc + 8]);
    const _Float16* gv = Vb + (size_t)vr * NS + kv0 + vc;
    cp16_g2l(gv,     &Vs[buf][vr][vc]);
    cp16_g2l(gv + 8, &Vs[buf][vr][vc + 8]);
  };

  // Q tile -> 4 persistent A fragments (16x32 each)
  half16_t qa[4];
  const _Float16* qrow = Qb + (size_t)(q0 + l16) * ND;
#pragma unroll
  for (int c = 0; c < 4; ++c) {
#pragma unroll
    for (int v = 0; v < 8; ++v) {
      int k = c * 32 + a_pair_base(v, hf);
      qa[c][2 * v]     = qrow[k];
      qa[c][2 * v + 1] = qrow[k + 1];
    }
  }

  const float8_t zero8 = {};
  float8_t o[8]; // O accumulator: 16 q-rows x 128 d, eight f32 C-fragments
  float m_r[8], l_r[8];
#pragma unroll
  for (int n = 0; n < 8; ++n) o[n] = zero8;
#pragma unroll
  for (int r = 0; r < 8; ++r) { m_r[r] = -1e30f; l_r[r] = 0.0f; }

  const float scale = 0.08838834764831843f; // 1/sqrt(128)

  stage(0, 0); // prologue fill of buffer 0
  int buf = 0;

  for (int kv0 = 0; kv0 < NS; kv0 += 32) {
    const bool more = (kv0 + 32 < NS);
    if (more) stage(buf ^ 1, kv0 + 32); // overlap next tile with this compute
#if HAS_ASYNC_LDS
    // current tile's 4 async ops done when only next tile's 4 remain
    if (more) __builtin_amdgcn_s_wait_asynccnt(4);
    else      __builtin_amdgcn_s_wait_asynccnt(0);
#else
    if (kv0 + 64 < NS) { // pull tile t+2 toward L2 on the fallback path
      __builtin_prefetch(Kb + (size_t)(kv0 + 64 + kr) * ND + kc, 0, 0);
      __builtin_prefetch(Vb + (size_t)vr * NS + kv0 + 64 + vc, 0, 0);
    }
#endif
    __syncthreads();

    // ---- scores: two 16x16 f32 tiles, 8 WMMAs over d=128 (K from LDS) ----
    float8_t s0 = zero8, s1 = zero8;
#pragma unroll
    for (int c = 0; c < 4; ++c) {
      s0 = wmma_f16(qa[c], load_b_contig(&Ks[buf][l16][c * 32 + hf * 16]), s0);
      s1 = wmma_f16(qa[c], load_b_contig(&Ks[buf][16 + l16][c * 32 + hf * 16]), s1);
    }

    // ---- |s|*scale, online softmax update, stage P to per-wave LDS ----
#pragma unroll
    for (int r = 0; r < 8; ++r) {
      float a0 = fabsf(s0[r]) * scale;
      float a1 = fabsf(s1[r]) * scale;
      float t = fmaxf(a0, a1); // row lives in one 16-lane half
      t = fmaxf(t, __shfl_xor(t, 1, 32));
      t = fmaxf(t, __shfl_xor(t, 2, 32));
      t = fmaxf(t, __shfl_xor(t, 4, 32));
      t = fmaxf(t, __shfl_xor(t, 8, 32));
      float mnew = fmaxf(m_r[r], t);
      float corr = __expf(m_r[r] - mnew);
      m_r[r] = mnew;
      float e0 = __expf(a0 - mnew);
      float e1 = __expf(a1 - mnew);
      float rs = e0 + e1;
      rs += __shfl_xor(rs, 1, 32);
      rs += __shfl_xor(rs, 2, 32);
      rs += __shfl_xor(rs, 4, 32);
      rs += __shfl_xor(rs, 8, 32);
      l_r[r] = l_r[r] * corr + rs;
      const int q = r + hf * 8; // C-layout row
      pl[q * 32 + l16]      = (_Float16)e0;
      pl[q * 32 + 16 + l16] = (_Float16)e1;
#pragma unroll
      for (int n = 0; n < 8; ++n) o[n][r] *= corr;
    }
    // per-wave LDS region; same-wave DS ops are in order (DScnt), compiler waits

    // ---- re-read P in A-fragment layout (16x32) ----
    half16_t pa;
#pragma unroll
    for (int v = 0; v < 8; ++v) {
      int k = a_pair_base(v, hf);
      pa[2 * v]     = pl[l16 * 32 + k];
      pa[2 * v + 1] = pl[l16 * 32 + k + 1];
    }

    // ---- O += P(16x32) x V(32x16) per 16-wide d chunk (V from LDS) ----
#pragma unroll
    for (int n = 0; n < 8; ++n)
      o[n] = wmma_f16(pa, load_b_contig(&Vs[buf][n * 16 + l16][hf * 16]), o[n]);

    __syncthreads(); // all waves done with this buffer before it is re-staged
    buf ^= 1;
  }

  // ---- epilogue: divide by softmax denominator, write f32 [B,S,H*DV] ----
#pragma unroll
  for (int r = 0; r < 8; ++r) {
    float inv = 1.0f / l_r[r];
    int q = q0 + r + hf * 8;
    size_t base = ((size_t)b * NS + q) * NN + (size_t)h * ND;
#pragma unroll
    for (int n = 0; n < 8; ++n)
      out[base + n * 16 + l16] = o[n][r] * inv;
  }
}

extern "C" void kernel_launch(void* const* d_in, const int* in_sizes, int n_in,
                              void* d_out, int out_size, void* d_ws, size_t ws_size,
                              hipStream_t stream) {
  (void)in_sizes; (void)n_in; (void)out_size; (void)ws_size;
  const float* query = (const float*)d_in[0];
  const float* key   = (const float*)d_in[1];
  const float* value = (const float*)d_in[2];
  const float* WQ = (const float*)d_in[3];
  const float* bQ = (const float*)d_in[4];
  const float* WK = (const float*)d_in[5];
  const float* bK = (const float*)d_in[6];
  const float* WV = (const float*)d_in[7];
  const float* bV = (const float*)d_in[8];

  const size_t elems = (size_t)NB * NH * NS * ND; // 8,388,608 f16 each
  _Float16* Qf = (_Float16*)d_ws;
  _Float16* Kf = Qf + elems;
  _Float16* Vt = Kf + elems; // total 48 MiB of workspace

  const int proj_blocks = ((NB * NS) / 16) * (NN / 16) / 4; // 8192 blocks x 4 waves
  proj_kernel<<<proj_blocks, 128, 0, stream>>>(query, WQ, bQ, Qf, 0);
  proj_kernel<<<proj_blocks, 128, 0, stream>>>(key,   WK, bK, Kf, 0);
  proj_kernel<<<proj_blocks, 128, 0, stream>>>(value, WV, bV, Vt, 1);

  const int attn_blocks = NB * NH * (NS / 128); // 512 blocks x 8 waves
  attn_kernel<<<attn_blocks, 256, 0, stream>>>(Qf, Kf, Vt, (float*)d_out);
}
